// GenePanelGNN_88630945120677
// MI455X (gfx1250) — matrix operand order repaired and checked by reading, not compiled, not run on hardware
//
#include <hip/hip_runtime.h>
#include <hip/hip_bf16.h>

// CDNA5 / gfx1250: wave32, WMMA 16x16x32 bf16 -> f32 accumulate.
typedef __attribute__((ext_vector_type(16))) __bf16 v16bf;
typedef __attribute__((ext_vector_type(8)))  __bf16 v8bf;
typedef __attribute__((ext_vector_type(8)))  float  v8f;

#define HIDX 128   // feature / hidden dim (d == HID == 128)

// ---------------------------------------------------------------------------
// B fragment from pre-transposed bf16 weights WT[n][k] (k contiguous).
// Lane L holds col N = L%16; elems 0..7 -> K = kb + half*8 + i,
// elems 8..15 -> K = kb + 16 + half*8 + i.  Two 16-byte loads per tile.
// ---------------------------------------------------------------------------
__device__ __forceinline__ v16bf load_bt(const __bf16* __restrict__ WT,
                                         int kb, int half, int ncol) {
  const v8bf* p = (const v8bf*)(WT + (size_t)ncol * HIDX + kb + half * 8);
  v8bf lo = p[0];
  v8bf hi = p[2];
  return __builtin_shufflevector(lo, hi, 0, 1, 2, 3, 4, 5, 6, 7,
                                 8, 9, 10, 11, 12, 13, 14, 15);
}

// ---------------------------------------------------------------------------
// A fragment from LDS-staged bf16 rows (row-major 16x128).
// Lane L holds row M = L%16; elems 0..7 -> K = kb + half*8 + i,
// elems 8..15 -> K = kb + 16 + half*8 + i.  Two ds_load_b128 per tile.
// ---------------------------------------------------------------------------
__device__ __forceinline__ v16bf load_a_lds(const __bf16* ldsrow,
                                            int kb, int half) {
  const v8bf* p0 = (const v8bf*)(ldsrow + kb + half * 8);
  const v8bf* p1 = (const v8bf*)(ldsrow + kb + 16 + half * 8);
  v8bf lo = *p0;
  v8bf hi = *p1;
  return __builtin_shufflevector(lo, hi, 0, 1, 2, 3, 4, 5, 6, 7,
                                 8, 9, 10, 11, 12, 13, 14, 15);
}

// A fragment straight from a global f32 row (bilinear gather path).
__device__ __forceinline__ v16bf load_a_glb(const float* __restrict__ rowp,
                                            int kb, int half) {
  v16bf a;
  const float* p0 = rowp + kb + half * 8;
  const float* p1 = rowp + kb + 16 + half * 8;
#pragma unroll
  for (int i = 0; i < 8; ++i) a[i] = (__bf16)p0[i];
#pragma unroll
  for (int i = 0; i < 8; ++i) a[8 + i] = (__bf16)p1[i];
  return a;
}

// ---------------------------------------------------------------------------
// One-off: transpose + convert the 9 weight matrices to bf16 WT[n][k].
// blockIdx.x selects the matrix.
// ---------------------------------------------------------------------------
__global__ void __launch_bounds__(256) prep_weights(
    const float* W0, const float* W1, const float* W2, const float* W3,
    const float* W4, const float* W5, const float* W6, const float* W7,
    const float* W8, __bf16* __restrict__ out) {
  const float* Ws[9] = {W0, W1, W2, W3, W4, W5, W6, W7, W8};
  const float* W = Ws[blockIdx.x];
  __bf16* o = out + (size_t)blockIdx.x * HIDX * HIDX;
  for (int i = threadIdx.x; i < HIDX * HIDX; i += 256) {
    int n = i >> 7;
    int k = i & 127;
    o[(size_t)n * HIDX + k] = (__bf16)W[(size_t)k * HIDX + n];
  }
}

// ---------------------------------------------------------------------------
// Degree counts for both directions (shared by both layers)
// ---------------------------------------------------------------------------
__global__ void __launch_bounds__(256) gnn_count_kernel(
    const int* __restrict__ src, const int* __restrict__ dst,
    float* __restrict__ cntG, float* __restrict__ cntP, int E) {
  int e = blockIdx.x * 256 + threadIdx.x;
  if (e >= E) return;
  atomicAdd(&cntP[dst[e]], 1.0f);
  atomicAdd(&cntG[src[e]], 1.0f);
}

// ---------------------------------------------------------------------------
// Bidirectional edge scatter: aggP[dst] += xsrc[src], aggG[src] += xdst[dst]
// One thread per (edge, 4-float chunk); float4 gathers + f32 atomics.
// ---------------------------------------------------------------------------
__global__ void __launch_bounds__(256) gnn_scatter_kernel(
    const float* __restrict__ xsrc, const float* __restrict__ xdst,
    const int* __restrict__ src, const int* __restrict__ dst,
    float* __restrict__ aggP, float* __restrict__ aggG, int E) {
  int tid = blockIdx.x * 256 + threadIdx.x;
  int e = tid >> 5;           // edge index
  int c = tid & 31;           // chunk of 4 floats (32 * 4 = 128)
  if (e >= E) return;
  int s = src[e];
  int d = dst[e];
  float4 vs = ((const float4*)(xsrc + (size_t)s * HIDX))[c];
  float4 vd = ((const float4*)(xdst + (size_t)d * HIDX))[c];
  float* ap = aggP + (size_t)d * HIDX + c * 4;
  float* ag = aggG + (size_t)s * HIDX + c * 4;
  atomicAdd(ap + 0, vs.x); atomicAdd(ap + 1, vs.y);
  atomicAdd(ap + 2, vs.z); atomicAdd(ap + 3, vs.w);
  atomicAdd(ag + 0, vd.x); atomicAdd(ag + 1, vd.y);
  atomicAdd(ag + 2, vd.z); atomicAdd(ag + 3, vd.w);
}

// ---------------------------------------------------------------------------
// Fused SAGE: out = [relu]( mean(agg,cnt) @ Wl + bl + xdst @ Wr )
// Block = 256 threads (8 waves). The block's 16 rows of mean/self features
// are staged once in LDS as bf16 (mean scale folded in); each wave owns one
// 16-column tile and issues 8 WMMAs (2 GEMMs x K=128/32).
// ---------------------------------------------------------------------------
__global__ void __launch_bounds__(256) sage_fused_wmma(
    const float* __restrict__ agg, const float* __restrict__ cnt,
    const float* __restrict__ xdst,
    const __bf16* __restrict__ WTl, const float* __restrict__ bl,
    const __bf16* __restrict__ WTr,
    float* __restrict__ out, int n, int do_relu) {
  __shared__ __bf16 ldsA[16 * HIDX];   // mean tile (scaled)
  __shared__ __bf16 ldsX[16 * HIDX];   // self tile
  const int rowbase = blockIdx.x * 16;

  {  // cooperative fill: thread -> (row, 8-col chunk)
    int r = threadIdx.x >> 4;               // 0..15
    int cc = (threadIdx.x & 15) * 8;        // 0..120
    int rowa = rowbase + r;
    if (rowa > n - 1) rowa = n - 1;         // clamp ragged tail
    float inv = 1.0f / fmaxf(cnt[rowa], 1.0f);
    const float* ar = agg  + (size_t)rowa * HIDX + cc;
    const float* xr = xdst + (size_t)rowa * HIDX + cc;
#pragma unroll
    for (int i = 0; i < 8; ++i) {
      ldsA[r * HIDX + cc + i] = (__bf16)(ar[i] * inv);
      ldsX[r * HIDX + cc + i] = (__bf16)xr[i];
    }
  }
  __syncthreads();

  const int lane = threadIdx.x & 31;
  const int wave = threadIdx.x >> 5;
  const int half = lane >> 4;
  const int lm   = lane & 15;
  const int ncol = wave * 16 + lm;
  const __bf16* arow = ldsA + lm * HIDX;
  const __bf16* xrow = ldsX + lm * HIDX;

  v8f c = {};
#pragma unroll
  for (int kb = 0; kb < HIDX; kb += 32) {
    v16bf a0 = load_a_lds(arow, kb, half);
    v16bf b0 = load_bt(WTl, kb, half, ncol);
    c = __builtin_amdgcn_wmma_f32_16x16x32_bf16(false, a0, false, b0,
                                                (short)0, c, false, false);
    v16bf a1 = load_a_lds(xrow, kb, half);
    v16bf b1 = load_bt(WTr, kb, half, ncol);
    c = __builtin_amdgcn_wmma_f32_16x16x32_bf16(false, a1, false, b1,
                                                (short)0, c, false, false);
  }
  const float bias = bl[ncol];
#pragma unroll
  for (int r = 0; r < 8; ++r) {                      // D: m = r + 8*half
    int orow = rowbase + half * 8 + r;
    if (orow < n) {
      float v = c[r] + bias;
      if (do_relu) v = fmaxf(v, 0.0f);
      out[(size_t)orow * HIDX + ncol] = v;
    }
  }
}

// ---------------------------------------------------------------------------
// Bilinear scorer: score[p] = g[p]^T Wb p[p] + bb
// One wave -> 16 pairs. T = G @ WbT via WMMA (8 column tiles x 4 K-steps),
// elementwise * gathered panel vectors, shfl_xor reduce over the 16-lane half.
// ---------------------------------------------------------------------------
__global__ void __launch_bounds__(256) bilinear_wmma(
    const float* __restrict__ oG, const float* __restrict__ oP,
    const int* __restrict__ pg, const int* __restrict__ pp,
    const __bf16* __restrict__ WbT, const float* __restrict__ bb,
    float* __restrict__ score, int P) {
  const int lane = threadIdx.x & 31;
  const int wave = threadIdx.x >> 5;
  const int half = lane >> 4;
  const int lm   = lane & 15;
  const int pbase = (blockIdx.x * 8 + wave) * 16;
  if (pbase >= P) return;                  // wave-uniform (P % 16 == 0)

  const int gi = pg[pbase + lm];
  const float* grow = oG + (size_t)gi * HIDX;
  v16bf a[4];
#pragma unroll
  for (int i = 0; i < 4; ++i) a[i] = load_a_glb(grow, i * 32, half);

  int ppi[8];
#pragma unroll
  for (int r = 0; r < 8; ++r) ppi[r] = pp[pbase + half * 8 + r];

  float acc[8] = {0.f, 0.f, 0.f, 0.f, 0.f, 0.f, 0.f, 0.f};
#pragma unroll
  for (int t = 0; t < 8; ++t) {
    const int nb = t * 16;
    v8f c = {};
#pragma unroll
    for (int i = 0; i < 4; ++i) {
      v16bf b = load_bt(WbT, i * 32, half, nb + lm);
      c = __builtin_amdgcn_wmma_f32_16x16x32_bf16(false, a[i], false, b,
                                                  (short)0, c, false, false);
    }
#pragma unroll
    for (int r = 0; r < 8; ++r)
      acc[r] += c[r] * oP[(size_t)ppi[r] * HIDX + nb + lm];
  }
  const float bias = bb[0];
#pragma unroll
  for (int r = 0; r < 8; ++r) {
    float v = acc[r];
    v += __shfl_xor(v, 1, 32);             // xor masks 1/2/4/8 stay within
    v += __shfl_xor(v, 2, 32);             // each 16-lane half -> sum over N
    v += __shfl_xor(v, 4, 32);
    v += __shfl_xor(v, 8, 32);
    if (lm == 0) score[pbase + half * 8 + r] = v + bias;
  }
}

// ---------------------------------------------------------------------------
// Host-side orchestration
// ---------------------------------------------------------------------------
extern "C" void kernel_launch(void* const* d_in, const int* in_sizes, int n_in,
                              void* d_out, int out_size, void* d_ws, size_t ws_size,
                              hipStream_t stream) {
  const float* x_gene  = (const float*)d_in[0];
  const float* x_panel = (const float*)d_in[1];
  const int*   e_src   = (const int*)d_in[2];
  const int*   e_dst   = (const int*)d_in[3];
  const int*   p_gene  = (const int*)d_in[4];
  const int*   p_panel = (const int*)d_in[5];
  const float* W1_gp_l = (const float*)d_in[6];
  const float* b1_gp   = (const float*)d_in[7];
  const float* W1_gp_r = (const float*)d_in[8];
  const float* W1_pg_l = (const float*)d_in[9];
  const float* b1_pg   = (const float*)d_in[10];
  const float* W1_pg_r = (const float*)d_in[11];
  const float* W2_gp_l = (const float*)d_in[12];
  const float* b2_gp   = (const float*)d_in[13];
  const float* W2_gp_r = (const float*)d_in[14];
  const float* W2_pg_l = (const float*)d_in[15];
  const float* b2_pg   = (const float*)d_in[16];
  const float* W2_pg_r = (const float*)d_in[17];
  const float* Wb      = (const float*)d_in[18];
  const float* bb      = (const float*)d_in[19];
  float* score = (float*)d_out;

  const int NG = in_sizes[0] / HIDX;   // 100000
  const int NP = in_sizes[1] / HIDX;   // 5000
  const int E  = in_sizes[2];          // 1000000
  const int P  = in_sizes[4];          // 200000

  // Workspace layout (floats), agg buffers reused across layers.
  float* ws = (float*)d_ws;
  const size_t szAggP = (size_t)NP * HIDX;
  const size_t szAggG = (size_t)NG * HIDX;
  const size_t szCntP = ((size_t)NP + 127) & ~(size_t)127;
  const size_t szCntG = ((size_t)NG + 127) & ~(size_t)127;
  float* aggP = ws;
  float* aggG = aggP + szAggP;
  float* cntP = aggG + szAggG;
  float* cntG = cntP + szCntP;
  float* hP   = cntG + szCntG;
  float* hG   = hP + szAggP;
  float* oP   = hG + szAggG;
  float* oG   = oP + szAggP;
  __bf16* wt  = (__bf16*)(oG + szAggG);   // 9 x 128x128 bf16 transposed weights
  const size_t WMAT = (size_t)HIDX * HIDX;

  const dim3 blk(256);
  const unsigned cntBlocks  = (unsigned)((E + 255) / 256);
  const unsigned scatBlocks = (unsigned)(((long long)E * 32 + 255) / 256);
  const unsigned tilesP     = (unsigned)((NP + 15) / 16);
  const unsigned tilesG     = (unsigned)((NG + 15) / 16);
  const unsigned pairBlocks = (unsigned)((P / 16 + 7) / 8);

  // --- Weight prep + zero accumulators ------------------------------------
  prep_weights<<<9, blk, 0, stream>>>(W1_gp_l, W1_gp_r, W1_pg_l, W1_pg_r,
                                      W2_gp_l, W2_gp_r, W2_pg_l, W2_pg_r,
                                      Wb, wt);
  hipMemsetAsync(aggP, 0, (szAggP + szAggG + szCntP + szCntG) * sizeof(float), stream);

  // --- Layer 1 ------------------------------------------------------------
  gnn_count_kernel<<<cntBlocks, blk, 0, stream>>>(e_src, e_dst, cntG, cntP, E);
  gnn_scatter_kernel<<<scatBlocks, blk, 0, stream>>>(x_gene, x_panel, e_src, e_dst,
                                                     aggP, aggG, E);
  sage_fused_wmma<<<tilesP, blk, 0, stream>>>(aggP, cntP, x_panel,
                                              wt + 0 * WMAT, b1_gp, wt + 1 * WMAT,
                                              hP, NP, 1);
  sage_fused_wmma<<<tilesG, blk, 0, stream>>>(aggG, cntG, x_gene,
                                              wt + 2 * WMAT, b1_pg, wt + 3 * WMAT,
                                              hG, NG, 1);

  // --- Layer 2 ------------------------------------------------------------
  hipMemsetAsync(aggP, 0, (szAggP + szAggG) * sizeof(float), stream);
  gnn_scatter_kernel<<<scatBlocks, blk, 0, stream>>>(hG, hP, e_src, e_dst,
                                                     aggP, aggG, E);
  sage_fused_wmma<<<tilesP, blk, 0, stream>>>(aggP, cntP, hP,
                                              wt + 4 * WMAT, b2_gp, wt + 5 * WMAT,
                                              oP, NP, 0);
  sage_fused_wmma<<<tilesG, blk, 0, stream>>>(aggG, cntG, hG,
                                              wt + 6 * WMAT, b2_pg, wt + 7 * WMAT,
                                              oG, NG, 0);

  // --- Bilinear scoring ----------------------------------------------------
  bilinear_wmma<<<pairBlocks, blk, 0, stream>>>(oG, oP, p_gene, p_panel,
                                                wt + 8 * WMAT, bb, score, P);
}
